// CvTAttentionBlock_52072183497281
// MI455X (gfx1250) — matrix operand order repaired and checked by reading, not compiled, not run on hardware
//
#include <hip/hip_runtime.h>

// ---------------------------------------------------------------------------
// CvT attention block for gfx1250 (MI455X), wave32 + WMMA f16 (16x16x32).
// Pipeline (all f16 intermediates, f32 accumulate):
//   1. dwconv3x3+BN (q: stride1, kv: stride2)        -> f16 token buffers
//   2. pointwise 192x192 GEMMs (WMMA), q scaled 1/8  -> qproj/kproj/vproj
//   3. build V'_i[k,c] = post[i,h(c)]*v[k,c], stored transposed [c][k], k padded to 800
//   4. attention pass 1: per-(b,i,q) softmax stats (m,l) via QK^T WMMAs + lane reductions
//   5. attention pass 2: normalized p (f16 via LDS relayout) x V'^T WMMAs -> out_tok f16
//   6. output projection GEMM (WMMA) -> f32 d_out
// ---------------------------------------------------------------------------

typedef __attribute__((ext_vector_type(16))) _Float16 v16h;
typedef __attribute__((ext_vector_type(8)))  _Float16 v8h;
typedef __attribute__((ext_vector_type(8)))  float    v8f;

#define DEVINL __device__ __forceinline__

constexpr int Bc  = 8;     // batch
constexpr int Lc  = 3136;  // q tokens (=56*56)
constexpr int Cc  = 192;   // channels
constexpr int S1c = 56;
constexpr int S2c = 28;
constexpr int LKc = 784;   // kv tokens (=28*28)
constexpr int LKP = 800;   // kv tokens padded to multiple of 32

DEVINL v8f wmma16(v16h a, v16h b, v8f c) {
  return __builtin_amdgcn_wmma_f32_16x16x32_f16(false, a, false, b, (short)0, c, false, false);
}

// A-operand (16x32 f16) per-lane load: elements 0..15 = K[g*16..g*16+15]
// (g = lane>>4), rows = lane&15.
DEVINL v16h load_a(const _Float16* row_chunk, int g) {
  union { v16h v; v8h h[2]; } u;
  u.h[0] = *(const v8h*)(row_chunk + g * 16);
  u.h[1] = *(const v8h*)(row_chunk + g * 16 + 8);
  return u.v;
}

// Row-wise reductions across the 16-lane half that holds one C-matrix row.
DEVINL float red_max16(float v) {
  v = fmaxf(v, __shfl_xor(v, 1));
  v = fmaxf(v, __shfl_xor(v, 2));
  v = fmaxf(v, __shfl_xor(v, 4));
  v = fmaxf(v, __shfl_xor(v, 8));
  return v;
}
DEVINL float red_sum16(float v) {
  v += __shfl_xor(v, 1);
  v += __shfl_xor(v, 2);
  v += __shfl_xor(v, 4);
  v += __shfl_xor(v, 8);
  return v;
}

// ---------------------------------------------------------------------------
// Weight transpose + f32->f16:  dst[f*192+c] = src[c*192+f]
__global__ void k_transpose_w(const float* __restrict__ src, _Float16* __restrict__ dst) {
  int idx = blockIdx.x * 256 + threadIdx.x;
  if (idx >= Cc * Cc) return;
  int f = idx / Cc, c = idx % Cc;
  dst[f * Cc + c] = (_Float16)src[c * Cc + f];
}

// ---------------------------------------------------------------------------
// Depthwise 3x3 stride-1 SAME (pad 1/1) + BN(eval) -> f16
__global__ void k_dwconv_s1(const float* __restrict__ x, const float* __restrict__ dw,
                            const float* __restrict__ sc, const float* __restrict__ bi,
                            const float* __restrict__ mu, const float* __restrict__ va,
                            _Float16* __restrict__ y) {
  int idx = blockIdx.x * 256 + threadIdx.x;
  if (idx >= Bc * Lc * Cc) return;
  int c  = idx % Cc;
  int x0 = (idx / Cc) % S1c;
  int y0 = (idx / (Cc * S1c)) % S1c;
  int b  = idx / (Cc * Lc);
  float s = 0.f;
#pragma unroll
  for (int dy = 0; dy < 3; ++dy) {
    int iy = y0 + dy - 1;
    if ((unsigned)iy >= (unsigned)S1c) continue;
#pragma unroll
    for (int dx = 0; dx < 3; ++dx) {
      int ix = x0 + dx - 1;
      if ((unsigned)ix >= (unsigned)S1c) continue;
      s += x[((size_t)b * Lc + iy * S1c + ix) * Cc + c] * dw[(dy * 3 + dx) * Cc + c];
    }
  }
  float a = sc[c] * rsqrtf(va[c] + 1e-5f);
  y[idx] = (_Float16)(s * a + (bi[c] - mu[c] * a));
}

// Depthwise 3x3 stride-2 SAME (pad 0/1) + BN for k AND v (shared input reads)
__global__ void k_dwconv_s2(const float* __restrict__ x,
                            const float* __restrict__ kdw, const float* __restrict__ ksc,
                            const float* __restrict__ kbi, const float* __restrict__ kmu,
                            const float* __restrict__ kva,
                            const float* __restrict__ vdw, const float* __restrict__ vsc,
                            const float* __restrict__ vbi, const float* __restrict__ vmu,
                            const float* __restrict__ vva,
                            _Float16* __restrict__ yk, _Float16* __restrict__ yv) {
  int idx = blockIdx.x * 256 + threadIdx.x;
  if (idx >= Bc * LKc * Cc) return;
  int c  = idx % Cc;
  int xo = (idx / Cc) % S2c;
  int yo = (idx / (Cc * S2c)) % S2c;
  int b  = idx / (Cc * LKc);
  float sk = 0.f, sv = 0.f;
#pragma unroll
  for (int dy = 0; dy < 3; ++dy) {
    int iy = yo * 2 + dy;          // pad_before = 0 for SAME stride2
    if (iy >= S1c) continue;
#pragma unroll
    for (int dx = 0; dx < 3; ++dx) {
      int ix = xo * 2 + dx;
      if (ix >= S1c) continue;
      float v = x[((size_t)b * Lc + iy * S1c + ix) * Cc + c];
      sk += v * kdw[(dy * 3 + dx) * Cc + c];
      sv += v * vdw[(dy * 3 + dx) * Cc + c];
    }
  }
  float ak = ksc[c] * rsqrtf(kva[c] + 1e-5f);
  float av = vsc[c] * rsqrtf(vva[c] + 1e-5f);
  yk[idx] = (_Float16)(sk * ak + (kbi[c] - kmu[c] * ak));
  yv[idx] = (_Float16)(sv * av + (vbi[c] - vmu[c] * av));
}

// ---------------------------------------------------------------------------
// Generic WMMA GEMM: C[M,192] = alpha * A[M,192] * Wt^T  (Wt stored [192n,192k])
// block = 128 threads (4 waves); wave -> 16 rows x full N=192 (12 tiles).
// Exactly one of Cf / Ch is non-null (f32 vs f16 output).
__global__ void k_gemm192(const _Float16* __restrict__ A, const _Float16* __restrict__ Wt,
                          float* __restrict__ Cf, _Float16* __restrict__ Ch, float alpha) {
  const int wave = threadIdx.x >> 5, lane = threadIdx.x & 31;
  const int lr = lane & 15, hi = lane >> 4;
  const size_t row0 = (size_t)blockIdx.x * 64 + wave * 16;
  v8f acc[12];
#pragma unroll
  for (int i = 0; i < 12; ++i) acc[i] = {};
  const _Float16* Ar = A + (row0 + lr) * Cc;
#pragma unroll
  for (int kc = 0; kc < 6; ++kc) {
    v16h a = load_a(Ar + kc * 32, hi);
#pragma unroll
    for (int nt = 0; nt < 12; ++nt) {
      v16h b = *(const v16h*)(Wt + (size_t)(nt * 16 + lr) * Cc + kc * 32 + hi * 16);
      acc[nt] = wmma16(a, b, acc[nt]);
    }
  }
#pragma unroll
  for (int nt = 0; nt < 12; ++nt)
#pragma unroll
    for (int j = 0; j < 8; ++j) {
      size_t off = (row0 + j + 8 * hi) * Cc + nt * 16 + lr;
      float v = acc[nt][j] * alpha;
      if (Cf) Cf[off] = v; else Ch[off] = (_Float16)v;
    }
}

// ---------------------------------------------------------------------------
// V'_i transposed + padded: vpT[b][i][c][k] = (k<784) ? post[i][c/64]*v[b][k][c] : 0
__global__ void k_vprime(const _Float16* __restrict__ vproj, const float* __restrict__ post,
                         _Float16* __restrict__ vpT) {
  int idx = blockIdx.x * 256 + threadIdx.x;
  if (idx >= Bc * 3 * Cc * LKP) return;
  int k = idx % LKP;
  int c = (idx / LKP) % Cc;
  int i = (idx / (LKP * Cc)) % 3;
  int b = idx / (LKP * Cc * 3);
  float v = 0.f;
  if (k < LKc) v = post[i * 3 + (c >> 6)] * (float)vproj[((size_t)b * LKc + k) * Cc + c];
  vpT[idx] = (_Float16)v;
}

// ---------------------------------------------------------------------------
// Attention pass 1: softmax stats. grid = (49, B), block = 128.
__global__ void k_att_ml(const _Float16* __restrict__ qb, const _Float16* __restrict__ kb,
                         const float* __restrict__ pre,
                         float* __restrict__ mbuf, float* __restrict__ lbuf) {
  const int b = blockIdx.y;
  const int wave = threadIdx.x >> 5, lane = threadIdx.x & 31;
  const int lr = lane & 15, hi = lane >> 4;
  const int q0 = blockIdx.x * 64 + wave * 16;

  float pm[3][3];
#pragma unroll
  for (int h = 0; h < 3; ++h)
#pragma unroll
    for (int i = 0; i < 3; ++i) pm[h][i] = pre[h * 3 + i];

  // Q operands: A-format, 3 heads x 2 K-chunks of 32 (d = 0..63); q already *0.125
  v16h aQ[3][2];
  const _Float16* qr = qb + ((size_t)b * Lc + q0 + lr) * Cc;
#pragma unroll
  for (int h = 0; h < 3; ++h)
#pragma unroll
    for (int cchunk = 0; cchunk < 2; ++cchunk)
      aQ[h][cchunk] = load_a(qr + h * 64 + cchunk * 32, hi);

  float m[3][8], l[3][8];
#pragma unroll
  for (int i = 0; i < 3; ++i)
#pragma unroll
    for (int j = 0; j < 8; ++j) { m[i][j] = -1e30f; l[i][j] = 0.f; }

  for (int ks = 0; ks < 25; ++ks) {
#pragma unroll
    for (int sub = 0; sub < 2; ++sub) {
      int k0 = ks * 32 + sub * 16;
      if (k0 < LKc) {
        v8f S[3];
#pragma unroll
        for (int h = 0; h < 3; ++h) {
          S[h] = {};
          const _Float16* kr = kb + ((size_t)b * LKc + k0 + lr) * Cc + h * 64 + hi * 16;
          v16h b0 = *(const v16h*)(kr);        // d 0..31 chunk (B-format)
          v16h b1 = *(const v16h*)(kr + 32);   // d 32..63 chunk
          S[h] = wmma16(aQ[h][0], b0, S[h]);
          S[h] = wmma16(aQ[h][1], b1, S[h]);
        }
#pragma unroll
        for (int i = 0; i < 3; ++i)
#pragma unroll
          for (int j = 0; j < 8; ++j) {
            float s  = pm[0][i] * S[0][j] + pm[1][i] * S[1][j] + pm[2][i] * S[2][j];
            float nm = fmaxf(m[i][j], red_max16(s));
            float rs = red_sum16(__expf(s - nm));
            l[i][j]  = l[i][j] * __expf(m[i][j] - nm) + rs;
            m[i][j]  = nm;
          }
      }
    }
  }
  if (lr == 0) {
#pragma unroll
    for (int i = 0; i < 3; ++i)
#pragma unroll
      for (int j = 0; j < 8; ++j) {
        size_t q = (size_t)q0 + j + 8 * hi;
        mbuf[((size_t)b * 3 + i) * Lc + q] = m[i][j];
        lbuf[((size_t)b * 3 + i) * Lc + q] = l[i][j];
      }
  }
}

// ---------------------------------------------------------------------------
// Attention pass 2: out_tok = sum_i softmax_i(Q K^T mix) * V'_i. grid = (49, B).
__global__ void k_att_out(const _Float16* __restrict__ qb, const _Float16* __restrict__ kb,
                          const _Float16* __restrict__ vpT, const float* __restrict__ pre,
                          const float* __restrict__ mbuf, const float* __restrict__ lbuf,
                          _Float16* __restrict__ outtok) {
  __shared__ __align__(128) _Float16 plds[4 * 3 * 512];  // 4 waves x 3 i x 16x32 f16
  const int b = blockIdx.y;
  const int wave = threadIdx.x >> 5, lane = threadIdx.x & 31;
  const int lr = lane & 15, hi = lane >> 4;
  const int q0 = blockIdx.x * 64 + wave * 16;
  _Float16* myl = &plds[wave * 3 * 512];

  float pm[3][3];
#pragma unroll
  for (int h = 0; h < 3; ++h)
#pragma unroll
    for (int i = 0; i < 3; ++i) pm[h][i] = pre[h * 3 + i];

  v16h aQ[3][2];
  const _Float16* qr = qb + ((size_t)b * Lc + q0 + lr) * Cc;
#pragma unroll
  for (int h = 0; h < 3; ++h)
#pragma unroll
    for (int cchunk = 0; cchunk < 2; ++cchunk)
      aQ[h][cchunk] = load_a(qr + h * 64 + cchunk * 32, hi);

  float m[3][8], linv[3][8];
#pragma unroll
  for (int i = 0; i < 3; ++i)
#pragma unroll
    for (int j = 0; j < 8; ++j) {
      size_t q = (size_t)q0 + j + 8 * hi;
      m[i][j]    = mbuf[((size_t)b * 3 + i) * Lc + q];
      linv[i][j] = 1.f / lbuf[((size_t)b * 3 + i) * Lc + q];
    }

  v8f acc[12];
#pragma unroll
  for (int i = 0; i < 12; ++i) acc[i] = {};

  for (int ks = 0; ks < 25; ++ks) {
    // produce normalized probability tiles (3 i's, 16q x 32k) into wave-private LDS
#pragma unroll
    for (int sub = 0; sub < 2; ++sub) {
      int k0 = ks * 32 + sub * 16;
      if (k0 < LKc) {
        v8f S[3];
#pragma unroll
        for (int h = 0; h < 3; ++h) {
          S[h] = {};
          const _Float16* kr = kb + ((size_t)b * LKc + k0 + lr) * Cc + h * 64 + hi * 16;
          v16h b0 = *(const v16h*)(kr);
          v16h b1 = *(const v16h*)(kr + 32);
          S[h] = wmma16(aQ[h][0], b0, S[h]);
          S[h] = wmma16(aQ[h][1], b1, S[h]);
        }
#pragma unroll
        for (int i = 0; i < 3; ++i)
#pragma unroll
          for (int j = 0; j < 8; ++j) {
            float s = pm[0][i] * S[0][j] + pm[1][i] * S[1][j] + pm[2][i] * S[2][j];
            float p = __expf(s - m[i][j]) * linv[i][j];
            myl[i * 512 + (j + 8 * hi) * 32 + sub * 16 + lr] = (_Float16)p;
          }
      } else {
#pragma unroll
        for (int i = 0; i < 3; ++i)
#pragma unroll
          for (int j = 0; j < 8; ++j)
            myl[i * 512 + (j + 8 * hi) * 32 + sub * 16 + lr] = (_Float16)0.f;
      }
    }
    __syncthreads();  // (regions are wave-private; barrier also orders DS ops)
    // PV' WMMAs: A = P_i (re-layout from LDS), B = V'^T rows
#pragma unroll
    for (int i = 0; i < 3; ++i) {
      v16h aP = load_a(myl + i * 512 + lr * 32, hi);
      const _Float16* vr = vpT + ((size_t)b * 3 + i) * Cc * LKP;
#pragma unroll
      for (int nt = 0; nt < 12; ++nt) {
        v16h bV = *(const v16h*)(vr + (size_t)(nt * 16 + lr) * LKP + ks * 32 + hi * 16);
        acc[nt] = wmma16(aP, bV, acc[nt]);
      }
    }
    __syncthreads();
  }

#pragma unroll
  for (int nt = 0; nt < 12; ++nt)
#pragma unroll
    for (int j = 0; j < 8; ++j)
      outtok[((size_t)b * Lc + q0 + j + 8 * hi) * Cc + nt * 16 + lr] = (_Float16)acc[nt][j];
}

// ---------------------------------------------------------------------------
extern "C" void kernel_launch(void* const* d_in, const int* in_sizes, int n_in,
                              void* d_out, int out_size, void* d_ws, size_t ws_size,
                              hipStream_t stream) {
  const float* inq  = (const float*)d_in[0];
  const float* inkv = (const float*)d_in[1];
  const float* q_dw = (const float*)d_in[2];
  const float* q_sc = (const float*)d_in[3];
  const float* q_bi = (const float*)d_in[4];
  const float* q_mu = (const float*)d_in[5];
  const float* q_va = (const float*)d_in[6];
  const float* q_pw = (const float*)d_in[7];
  const float* k_dw = (const float*)d_in[8];
  const float* k_sc = (const float*)d_in[9];
  const float* k_bi = (const float*)d_in[10];
  const float* k_mu = (const float*)d_in[11];
  const float* k_va = (const float*)d_in[12];
  const float* k_pw = (const float*)d_in[13];
  const float* v_dw = (const float*)d_in[14];
  const float* v_sc = (const float*)d_in[15];
  const float* v_bi = (const float*)d_in[16];
  const float* v_mu = (const float*)d_in[17];
  const float* v_va = (const float*)d_in[18];
  const float* v_pw = (const float*)d_in[19];
  const float* pre  = (const float*)d_in[20];
  const float* post = (const float*)d_in[21];
  const float* wout = (const float*)d_in[22];

  // workspace carve (256B aligned slabs), ~45 MB total
  char* p = (char*)d_ws;
  auto take = [&](size_t bytes) -> char* {
    char* r = p;
    p += (bytes + 255) & ~(size_t)255;
    return r;
  };
  _Float16* wqT    = (_Float16*)take((size_t)Cc * Cc * 2);
  _Float16* wkT    = (_Float16*)take((size_t)Cc * Cc * 2);
  _Float16* wvT    = (_Float16*)take((size_t)Cc * Cc * 2);
  _Float16* woT    = (_Float16*)take((size_t)Cc * Cc * 2);
  _Float16* dwq    = (_Float16*)take((size_t)Bc * Lc * Cc * 2);
  _Float16* dwk    = (_Float16*)take((size_t)Bc * LKc * Cc * 2);
  _Float16* dwv    = (_Float16*)take((size_t)Bc * LKc * Cc * 2);
  _Float16* qproj  = (_Float16*)take((size_t)Bc * Lc * Cc * 2);
  _Float16* kproj  = (_Float16*)take((size_t)Bc * LKc * Cc * 2);
  _Float16* vproj  = (_Float16*)take((size_t)Bc * LKc * Cc * 2);
  _Float16* vpT    = (_Float16*)take((size_t)Bc * 3 * Cc * LKP * 2);
  _Float16* outtok = (_Float16*)take((size_t)Bc * Lc * Cc * 2);
  float*    mbuf   = (float*)take((size_t)Bc * 3 * Lc * 4);
  float*    lbuf   = (float*)take((size_t)Bc * 3 * Lc * 4);

  // 1) weight transposes (f32 -> f16)
  k_transpose_w<<<(Cc * Cc + 255) / 256, 256, 0, stream>>>(q_pw, wqT);
  k_transpose_w<<<(Cc * Cc + 255) / 256, 256, 0, stream>>>(k_pw, wkT);
  k_transpose_w<<<(Cc * Cc + 255) / 256, 256, 0, stream>>>(v_pw, wvT);
  k_transpose_w<<<(Cc * Cc + 255) / 256, 256, 0, stream>>>(wout, woT);

  // 2) depthwise conv + BN
  k_dwconv_s1<<<(Bc * Lc * Cc + 255) / 256, 256, 0, stream>>>(
      inq, q_dw, q_sc, q_bi, q_mu, q_va, dwq);
  k_dwconv_s2<<<(Bc * LKc * Cc + 255) / 256, 256, 0, stream>>>(
      inkv, k_dw, k_sc, k_bi, k_mu, k_va, v_dw, v_sc, v_bi, v_mu, v_va, dwk, dwv);

  // 3) pointwise projections (WMMA GEMMs); q scaled by 1/sqrt(64) = 0.125
  k_gemm192<<<(Bc * Lc) / 64, 128, 0, stream>>>(dwq, wqT, nullptr, qproj, 0.125f);
  k_gemm192<<<(Bc * LKc) / 64, 128, 0, stream>>>(dwk, wkT, nullptr, kproj, 1.0f);
  k_gemm192<<<(Bc * LKc) / 64, 128, 0, stream>>>(dwv, wvT, nullptr, vproj, 1.0f);

  // 4) post-softmax head mix folded into V (transposed, k padded to 800)
  k_vprime<<<(Bc * 3 * Cc * LKP + 255) / 256, 256, 0, stream>>>(vproj, post, vpT);

  // 5) attention: stats pass then output pass
  k_att_ml<<<dim3(Lc / 64, Bc), 128, 0, stream>>>(qproj, kproj, pre, mbuf, lbuf);
  k_att_out<<<dim3(Lc / 64, Bc), 128, 0, stream>>>(qproj, kproj, vpT, pre, mbuf, lbuf, outtok);

  // 6) output projection -> f32 d_out
  k_gemm192<<<(Bc * Lc) / 64, 128, 0, stream>>>(outtok, woT, (float*)d_out, nullptr, 1.0f);
}
